// MultiHeadAttention_40183714022019
// MI455X (gfx1250) — matrix operand order repaired and checked by reading, not compiled, not run on hardware
//
#include <hip/hip_runtime.h>
#include <stdint.h>

typedef __attribute__((ext_vector_type(16))) __bf16 v16bf;
typedef __attribute__((ext_vector_type(8)))  float  v8f;
typedef __attribute__((ext_vector_type(4)))  unsigned int u32x4;
typedef __attribute__((ext_vector_type(2)))  unsigned int u32x2;
typedef __attribute__((ext_vector_type(4)))  float  f32x4;

#define BM 128
#define BN 128
#define BK 32
#define LDA_S 40   // padded row stride (elements) for LDS A tile (80 B)
#define LDB_S 40   // padded row stride (elements) for LDS B^T tile (80 B)

__host__ __device__ static inline unsigned short bfbits_f(float f) {
  union { float f; unsigned int u; } x; x.f = f;
  unsigned int r = (x.u + 0x7FFFu + ((x.u >> 16) & 1u)) >> 16;
  return (unsigned short)r;
}
__host__ __device__ static inline unsigned short bfbits_b(__bf16 b) {
  unsigned short h; __builtin_memcpy(&h, &b, 2); return h;
}
__host__ __device__ static inline __bf16 f2bf(float f) {
  unsigned short h = bfbits_f(f);
  __bf16 b; __builtin_memcpy(&b, &h, 2); return b;
}
__device__ static inline unsigned pack2(unsigned short lo, unsigned short hi) {
  return (unsigned)lo | ((unsigned)hi << 16);
}
__device__ static inline unsigned lds_off_u32(const void* p) {
  // low 32 bits of a generic pointer into LDS = byte offset within the
  // wave's LDS allocation (aperture bits live in the high half).
  return (unsigned)(uintptr_t)p;
}

// Generic batched GEMM:  C = alpha * (A @ B + bias)
//   A: [M,K] row-major, fp32 (Af) or bf16 (Ab); row stride lda
//   B: if bTrans==0: [K,N] row-major (row stride ldb)
//      if bTrans==1: B^T given as [N,K] row-major (row stride ldb)
//   C: fp32 (Cf) and/or bf16 (Ch); row stride ldc
//   batch: z = bz*nh + hz; per-operand offsets bz*s?b + hz*s?h
// bf16 operand staging uses GLOBAL_LOAD_ASYNC_TO_LDS_B128 (ASYNCcnt);
// fp32 operands are converted to bf16 through VGPRs with packed ds stores.
__global__ __launch_bounds__(256) void gemm_bf16_wmma(
    const float* __restrict__ Af, const __bf16* __restrict__ Ab, long lda,
    const float* __restrict__ Bf, const __bf16* __restrict__ Bb, long ldb, int bTrans,
    const float* __restrict__ bias, float alpha,
    float* __restrict__ Cf, __bf16* __restrict__ Ch, long ldc,
    int M, int N, int K,
    int nh, long sAb, long sAh, long sBb, long sBh, long sCb, long sCh)
{
  __shared__ alignas(16) __bf16 sA[BM * LDA_S];
  __shared__ alignas(16) __bf16 sB[BN * LDB_S];

  const int tid   = threadIdx.x;
  const int lane  = tid & 31;
  const int wave  = tid >> 5;
  const int mwave = (wave & 3) * 32;   // 4 wave-rows of 32
  const int nwave = (wave >> 2) * 64;  // 2 wave-cols of 64
  const int rowA  = lane & 15;         // M (A frag) / N (B frag, C) index
  const int kg    = lane >> 4;         // lane-half selector

  const int z  = blockIdx.z;
  const int bz = z / nh;
  const int hz = z - bz * nh;
  const long offA = (long)bz * sAb + (long)hz * sAh;
  const long offB = (long)bz * sBb + (long)hz * sBh;
  const long offC = (long)bz * sCb + (long)hz * sCh;

  const int m0 = blockIdx.y * BM;
  const int n0 = blockIdx.x * BN;

  const bool asyncA = (Af == nullptr);                 // bf16 A: direct async copy
  const bool asyncB = (bTrans != 0) && (Bf == nullptr); // bf16 B^T: direct async copy

  v8f acc[2][4] = {};

  for (int k0 = 0; k0 < K; k0 += BK) {
    __syncthreads();

    // ---- stage A tile: BM x BK ----
    if (asyncA) {
      // 16 B per lane per op, 2 ops: async global -> LDS, no VGPR data path
      const int rr0 = tid >> 2;          // 0..63
      const int kc8 = (tid & 3) * 8;     // 0,8,16,24 (x2B = 16B aligned)
      #pragma unroll
      for (int p = 0; p < 2; ++p) {
        const int rr = rr0 + p * 64;
        const unsigned long long ga =
            (unsigned long long)(const void*)(Ab + offA + (long)(m0 + rr) * lda + (k0 + kc8));
        const unsigned ldsd = lds_off_u32(&sA[rr * LDA_S + kc8]);
        asm volatile("global_load_async_to_lds_b128 %0, %1, off"
                     :: "v"(ldsd), "v"(ga) : "memory");
      }
    } else {
      // fp32 -> bf16 convert through VGPRs; b128 load, packed b64 LDS store
      const int r  = tid >> 3;        // 0..31
      const int kc = (tid & 7) * 4;   // 0..28
      #pragma unroll
      for (int p = 0; p < 4; ++p) {
        const int rr = r + p * 32;    // M always a multiple of BM
        const long ga = offA + (long)(m0 + rr) * lda + (k0 + kc);
        if (k0 + BK < K) __builtin_prefetch(Af + ga + BK, 0, 1);
        const f32x4 v = *(const f32x4*)(Af + ga);
        u32x2 w;
        w.x = pack2(bfbits_f(v.x), bfbits_f(v.y));
        w.y = pack2(bfbits_f(v.z), bfbits_f(v.w));
        *(u32x2*)&sA[rr * LDA_S + kc] = w;
      }
    }

    // ---- stage B tile into transposed LDS layout sB[n][k] ----
    if (bTrans) {
      if (asyncB) {
        const int nn0 = tid >> 2;
        const int kc8 = (tid & 3) * 8;
        #pragma unroll
        for (int p = 0; p < 2; ++p) {
          const int nn = nn0 + p * 64;
          if ((n0 + nn) < N) {
            const unsigned long long gb =
                (unsigned long long)(const void*)(Bb + offB + (long)(n0 + nn) * ldb + (k0 + kc8));
            const unsigned ldsd = lds_off_u32(&sB[nn * LDB_S + kc8]);
            asm volatile("global_load_async_to_lds_b128 %0, %1, off"
                         :: "v"(ldsd), "v"(gb) : "memory");
          } else {
            *(u32x4*)&sB[nn * LDB_S + kc8] = (u32x4){0u, 0u, 0u, 0u};
          }
        }
      } else {
        // fp32 B^T: convert through VGPRs, packed b64 stores
        const int r  = tid >> 3;
        const int kc = (tid & 7) * 4;
        #pragma unroll
        for (int p = 0; p < 4; ++p) {
          const int nn = r + p * 32;
          const bool ok = (n0 + nn) < N;
          const long gb = offB + (long)(n0 + nn) * ldb + (k0 + kc);
          u32x2 w = {0u, 0u};
          if (ok) {
            const f32x4 v = *(const f32x4*)(Bf + gb);
            w.x = pack2(bfbits_f(v.x), bfbits_f(v.y));
            w.y = pack2(bfbits_f(v.z), bfbits_f(v.w));
          }
          *(u32x2*)&sB[nn * LDB_S + kc] = w;
        }
      }
    } else {
      // B given as [K,N]: 2x2 micro-tile transpose, packed u32 LDS stores
      const int k2 = (tid >> 6) * 2;      // 0,2,4,6
      const int nn = (tid & 63) * 2;      // 0..126
      #pragma unroll
      for (int p = 0; p < 4; ++p) {
        const int k = k2 + p * 8;         // even k in 0..30
        const long g0 = offB + (long)(k0 + k) * ldb + (n0 + nn);
        const long g1 = g0 + ldb;
        const bool ok0 = (n0 + nn) < N;
        const bool ok1 = (n0 + nn + 1) < N;
        unsigned short b00 = 0, b01 = 0, b10 = 0, b11 = 0;
        if (Bf) {
          if (ok0) { b00 = bfbits_f(Bf[g0]);     b10 = bfbits_f(Bf[g1]); }
          if (ok1) { b01 = bfbits_f(Bf[g0 + 1]); b11 = bfbits_f(Bf[g1 + 1]); }
        } else {
          if (ok0) { b00 = bfbits_b(Bb[g0]);     b10 = bfbits_b(Bb[g1]); }
          if (ok1) { b01 = bfbits_b(Bb[g0 + 1]); b11 = bfbits_b(Bb[g1 + 1]); }
        }
        *(unsigned*)&sB[nn * LDB_S + k]       = pack2(b00, b10);
        *(unsigned*)&sB[(nn + 1) * LDB_S + k] = pack2(b01, b11);
      }
    }

    if (asyncA || asyncB)
      asm volatile("s_wait_asynccnt 0x0" ::: "memory");
    __syncthreads();

    // ---- load fragments (ds_load_b128 pairs) and issue 8 WMMAs ----
    v16bf afr[2], bfr[4];
    #pragma unroll
    for (int i = 0; i < 2; ++i) {
      // A 16-bit layout: lanes 0-15 hold K{0..7,16..23}, lanes 16-31 K{8..15,24..31}
      const __bf16* pa = &sA[(mwave + i * 16 + rowA) * LDA_S + kg * 8];
      u32x4* d = (u32x4*)&afr[i];
      d[0] = *(const u32x4*)(pa);
      d[1] = *(const u32x4*)(pa + 16);
    }
    #pragma unroll
    for (int j = 0; j < 4; ++j) {
      // B 16-bit layout: lanes 0-15 hold K{0..7,16..23}, lanes 16-31 K{8..15,24..31}
      const __bf16* pb = &sB[(nwave + j * 16 + rowA) * LDB_S + kg * 8];
      u32x4* d = (u32x4*)&bfr[j];
      d[0] = *(const u32x4*)(pb);
      d[1] = *(const u32x4*)(pb + 16);
    }
    #pragma unroll
    for (int i = 0; i < 2; ++i)
      #pragma unroll
      for (int j = 0; j < 4; ++j)
        acc[i][j] = __builtin_amdgcn_wmma_f32_16x16x32_bf16(
            false, afr[i], false, bfr[j], (short)0, acc[i][j], false, false);
  }

  // ---- epilogue: C 16x16 f32 layout: VGPR r -> M = r + 8*(lane>>4), N = lane&15
  for (int i = 0; i < 2; ++i) {
    const int mrow = m0 + mwave + i * 16 + kg * 8;
    for (int j = 0; j < 4; ++j) {
      const int nc = n0 + nwave + j * 16 + rowA;
      if (nc < N) {
        const float badd = bias ? bias[nc] : 0.0f;
        #pragma unroll
        for (int r = 0; r < 8; ++r) {
          const float v = (acc[i][j][r] + badd) * alpha;
          const long idx = offC + (long)(mrow + r) * ldc + nc;
          if (Cf) Cf[idx] = v;
          if (Ch) Ch[idx] = f2bf(v);
        }
      }
    }
  }
}

// In-place numerically-stable row softmax: one block per row.
__global__ __launch_bounds__(256) void softmax_rows(float* __restrict__ w, int cols) {
  __shared__ float red[256];
  const long row = blockIdx.x;
  float* p = w + row * (long)cols;
  const int tid = threadIdx.x;

  float m = -3.4e38f;
  for (int c = tid; c < cols; c += 256) m = fmaxf(m, p[c]);
  red[tid] = m; __syncthreads();
  for (int s = 128; s > 0; s >>= 1) {
    if (tid < s) red[tid] = fmaxf(red[tid], red[tid + s]);
    __syncthreads();
  }
  m = red[0]; __syncthreads();

  float sum = 0.0f;
  for (int c = tid; c < cols; c += 256) {
    const float e = expf(p[c] - m);
    p[c] = e;
    sum += e;
  }
  red[tid] = sum; __syncthreads();
  for (int s = 128; s > 0; s >>= 1) {
    if (tid < s) red[tid] += red[tid + s];
    __syncthreads();
  }
  const float inv = 1.0f / red[0];
  for (int c = tid; c < cols; c += 256) p[c] *= inv;
}

extern "C" void kernel_launch(void* const* d_in, const int* in_sizes, int n_in,
                              void* d_out, int out_size, void* d_ws, size_t ws_size,
                              hipStream_t stream) {
  (void)in_sizes; (void)n_in; (void)out_size; (void)ws_size;

  const int Bsz = 2, S = 2048, DMv = 1024, Hh = 16, Dd = 64;
  const long SD = (long)S * DMv;   // per-batch stride of [S,DM] tensors
  const long SS = (long)S * S;     // per-head stride of [S,S] weights

  const float* Q  = (const float*)d_in[0];
  const float* Km = (const float*)d_in[1];
  const float* V  = (const float*)d_in[2];
  const float* Wq = (const float*)d_in[3];
  const float* bq = (const float*)d_in[4];
  const float* bk = (const float*)d_in[6];
  const float* Wk = (const float*)d_in[5];
  const float* Wv = (const float*)d_in[7];
  const float* bv = (const float*)d_in[8];
  const float* Wo = (const float*)d_in[9];
  const float* bo = (const float*)d_in[10];

  float* out = (float*)d_out;                 // [2,2048,1024]
  float* wts = out + (long)Bsz * SD;          // [2,16,2048,2048]

  // ws layout (bf16): q | k | v | concat, each B*S*DM elements (8 MB) = 32 MB total
  __bf16* qb = (__bf16*)d_ws;
  __bf16* kb = qb + (long)Bsz * SD;
  __bf16* vb = kb + (long)Bsz * SD;
  __bf16* cb = vb + (long)Bsz * SD;

  const dim3 blk(256, 1, 1);
  const int M1 = Bsz * S;  // 4096 (batch folded into M for projections)

  // 1-3) Projections: fp32 X @ fp32 W + b -> bf16 (Q scaled by 1/sqrt(D))
  const dim3 gp(DMv / BN, M1 / BM, 1);
  gemm_bf16_wmma<<<gp, blk, 0, stream>>>(Q, nullptr, DMv, Wq, nullptr, DMv, 0,
      bq, 0.125f, nullptr, qb, DMv, M1, DMv, DMv, 1, 0, 0, 0, 0, 0, 0);
  gemm_bf16_wmma<<<gp, blk, 0, stream>>>(Km, nullptr, DMv, Wk, nullptr, DMv, 0,
      bk, 1.0f, nullptr, kb, DMv, M1, DMv, DMv, 1, 0, 0, 0, 0, 0, 0);
  gemm_bf16_wmma<<<gp, blk, 0, stream>>>(V, nullptr, DMv, Wv, nullptr, DMv, 0,
      bv, 1.0f, nullptr, vb, DMv, M1, DMv, DMv, 1, 0, 0, 0, 0, 0, 0);

  // 4) logits[b,h] = q_s @ k^T  (both operands staged via async global->LDS)
  const dim3 gl(S / BN, S / BM, Bsz * Hh);
  gemm_bf16_wmma<<<gl, blk, 0, stream>>>(nullptr, qb, DMv, nullptr, kb, DMv, 1,
      nullptr, 1.0f, wts, nullptr, S,
      S, S, Dd,
      Hh, SD, (long)Dd, SD, (long)Dd, (long)Hh * SS, SS);

  // 5) softmax in place over each of B*H*S rows
  softmax_rows<<<dim3((unsigned)(Bsz * Hh * S), 1, 1), blk, 0, stream>>>(wts, S);

  // 6) attn[b,h] = P(fp32, converted on the fly) @ v -> bf16 concat [B,S,DM]
  const dim3 ga((Dd + BN - 1) / BN, S / BM, Bsz * Hh);
  gemm_bf16_wmma<<<ga, blk, 0, stream>>>(wts, nullptr, S, nullptr, vb, DMv, 0,
      nullptr, 1.0f, nullptr, cb, DMv,
      S, Dd, S,
      Hh, (long)Hh * SS, SS, SD, (long)Dd, SD, (long)Dd);

  // 7) output = concat @ Wo + bo -> fp32 output region (A staged async)
  gemm_bf16_wmma<<<gp, blk, 0, stream>>>(nullptr, cb, DMv, Wo, nullptr, DMv, 0,
      bo, 1.0f, out, nullptr, DMv, M1, DMv, DMv, 1, 0, 0, 0, 0, 0, 0);
}
